// kv_cache_group_query_58686433133032
// MI455X (gfx1250) — matrix operand order repaired and checked by reading, not compiled, not run on hardware
//
#include <hip/hip_runtime.h>
#include <hip/hip_bf16.h>
#include <math.h>

#define EMB    4096
#define QH     32
#define KVH    8
#define HD     128
#define GQ     4          // QH / KVH
#define BB     2
#define SS     1024
#define KV_LEN 4096
#define START  1024
#define TT     2048       // START + SS
#define NK     (KVH * HD) // 1024
#define MROWS  (BB * SS)  // 2048

typedef __attribute__((ext_vector_type(16))) __bf16        v16bf;
typedef __attribute__((ext_vector_type(8)))  float         v8f;
typedef __attribute__((ext_vector_type(4)))  unsigned int  v4u;

union Frag {
    v16bf v;
    v4u   u[2];
};

__device__ __forceinline__ unsigned short f2bf(float f) {
    unsigned int u = __float_as_uint(f);
    unsigned int r = u + 0x7FFFu + ((u >> 16) & 1u);   // round-to-nearest-even
    return (unsigned short)(r >> 16);
}

// A-fragment load: lane row fixed, k = hi8 + {0..7, 16..23}
#define LDA(fr, rowptr, kkv)                                                \
    do {                                                                    \
        (fr).u[0] = *(const v4u*)((rowptr) + (kkv) + hiA);                  \
        (fr).u[1] = *(const v4u*)((rowptr) + (kkv) + hiA + 16);             \
    } while (0)
// B-fragment load: lane col fixed, k = hi8 + {0..7, 16..23}
#define LDB(fr, rowptr, kkv)                                                \
    do {                                                                    \
        (fr).u[0] = *(const v4u*)((rowptr) + (kkv) + hiA);                  \
        (fr).u[1] = *(const v4u*)((rowptr) + (kkv) + hiA + 16);             \
    } while (0)

// ---------------------------------------------------------------- converts
__global__ void f32_to_bf16_kernel(const float* __restrict__ src,
                                   unsigned short* __restrict__ dst, int n) {
    int i = blockIdx.x * blockDim.x + threadIdx.x;
    if (i < n) dst[i] = f2bf(src[i]);
}

// W (K x N fp32, row-major) -> Wt (N x K bf16, row-major)
__global__ void transpose_to_bf16_kernel(const float* __restrict__ W,
                                         unsigned short* __restrict__ Wt,
                                         int K, int N) {
    int i = blockIdx.x * blockDim.x + threadIdx.x;
    if (i >= K * N) return;
    int k = i / N;
    int n = i - k * N;
    Wt[(size_t)n * K + k] = f2bf(W[i]);
}

// ---------------------------------------------------------------- GEMM
// C[M x N] (fp32) = A[M x K] (bf16) * Bt[N x K]^T (bf16)  (+ bias[n] if given)
// wave tile: 32(M) x 64(N); block: 8 waves -> 64(M) x 256(N)
// K must be a multiple of 64 (always 4096 here).
__global__ void __launch_bounds__(256)
gemm_bf16_wmma_kernel(const unsigned short* __restrict__ A,
                      const unsigned short* __restrict__ Bt,
                      float* __restrict__ C,
                      const float* __restrict__ bias,
                      int M, int N, int K) {
    const int lane = threadIdx.x & 31;
    const int wave = threadIdx.x >> 5;
    const int wm   = wave >> 2;
    const int wn   = wave & 3;
    const int m0   = blockIdx.y * 64 + wm * 32;
    const int n0   = blockIdx.x * 256 + wn * 64;
    const int l15  = lane & 15;
    const int hiA  = (lane >> 4) * 8;    // A k-offset: 0 / 8
    const int hiB  = (lane >> 4) * 16;   // unused (B loads use hiA)

    v8f acc[2][4];
#pragma unroll
    for (int h = 0; h < 2; ++h)
#pragma unroll
        for (int j = 0; j < 4; ++j)
#pragma unroll
            for (int r = 0; r < 8; ++r) acc[h][j][r] = 0.0f;

    const unsigned short* arow0 = A + (size_t)(m0 + l15) * K;
    const unsigned short* arow1 = A + (size_t)(m0 + 16 + l15) * K;
    const unsigned short* brow[4];
#pragma unroll
    for (int j = 0; j < 4; ++j)
        brow[j] = Bt + (size_t)(n0 + j * 16 + l15) * K;

    // double-buffered fragments (distinct registers -> loads overlap WMMAs)
    Frag a0A, a1A, bA[4];
    Frag a0B, a1B, bB[4];

    LDA(a0A, arow0, 0);
    LDA(a1A, arow1, 0);
#pragma unroll
    for (int j = 0; j < 4; ++j) LDB(bA[j], brow[j], 0);

    for (int kk = 0; kk < K; kk += 64) {
        const int k1 = kk + 32;
        LDA(a0B, arow0, k1);
        LDA(a1B, arow1, k1);
#pragma unroll
        for (int j = 0; j < 4; ++j) LDB(bB[j], brow[j], k1);

        // prefetch ~1KB ahead (global_prefetch_b8); uniform guard
        const int kp = kk + 512;
        if (kp < K) {
            __builtin_prefetch(arow0 + kp, 0, 3);
            __builtin_prefetch(arow1 + kp, 0, 3);
#pragma unroll
            for (int j = 0; j < 4; ++j) __builtin_prefetch(brow[j] + kp, 0, 3);
        }

#pragma unroll
        for (int j = 0; j < 4; ++j) {
            acc[0][j] = __builtin_amdgcn_wmma_f32_16x16x32_bf16(
                false, a0A.v, false, bA[j].v, (short)0, acc[0][j], false, false);
            acc[1][j] = __builtin_amdgcn_wmma_f32_16x16x32_bf16(
                false, a1A.v, false, bA[j].v, (short)0, acc[1][j], false, false);
        }

        const int k2 = (kk + 64 < K) ? kk + 64 : 0;   // tail wraps (harmless)
        LDA(a0A, arow0, k2);
        LDA(a1A, arow1, k2);
#pragma unroll
        for (int j = 0; j < 4; ++j) LDB(bA[j], brow[j], k2);

#pragma unroll
        for (int j = 0; j < 4; ++j) {
            acc[0][j] = __builtin_amdgcn_wmma_f32_16x16x32_bf16(
                false, a0B.v, false, bB[j].v, (short)0, acc[0][j], false, false);
            acc[1][j] = __builtin_amdgcn_wmma_f32_16x16x32_bf16(
                false, a1B.v, false, bB[j].v, (short)0, acc[1][j], false, false);
        }
    }

#pragma unroll
    for (int h = 0; h < 2; ++h)
#pragma unroll
        for (int j = 0; j < 4; ++j) {
            const int n = n0 + j * 16 + l15;
            const float bb = bias ? bias[n] : 0.0f;
#pragma unroll
            for (int r = 0; r < 8; ++r) {
                const int m = m0 + h * 16 + hiA + r;  // C-layout row
                C[(size_t)m * N + n] = acc[h][j][r] + bb;
            }
        }
}

// ---------------------------------------------------------------- RoPE / cache build
__global__ void rope_q_kernel(const float* __restrict__ qf,
                              unsigned short* __restrict__ Qbf) {
    int i = blockIdx.x * blockDim.x + threadIdx.x;     // pair index
    const int total = BB * SS * QH * (HD / 2);
    if (i >= total) return;
    int p = i % (HD / 2);
    int h = (i / (HD / 2)) % QH;
    int s = (i / ((HD / 2) * QH)) % SS;
    int b = i / ((HD / 2) * QH * SS);
    const float* row = qf + (size_t)(b * SS + s) * EMB + h * HD;
    float xr = row[2 * p], xi = row[2 * p + 1];
    float inv = powf(10000.0f, -(float)(2 * p) / (float)HD);
    float ang = (float)s * inv;
    float sn, cs;
    sincosf(ang, &sn, &cs);
    unsigned short* orow = Qbf + (size_t)((b * QH + h) * SS + s) * HD;
    orow[2 * p]     = f2bf(xr * cs - xi * sn);
    orow[2 * p + 1] = f2bf(xr * sn + xi * cs);
}

__global__ void build_k_kernel(const float* __restrict__ cache_k,
                               const float* __restrict__ kf,
                               unsigned short* __restrict__ Kbf) {
    int i = blockIdx.x * blockDim.x + threadIdx.x;     // pair index
    const int total = BB * KVH * TT * (HD / 2);
    if (i >= total) return;
    int p  = i % (HD / 2);
    int t  = (i / (HD / 2)) % TT;
    int kh = (i / ((HD / 2) * TT)) % KVH;
    int b  = i / ((HD / 2) * TT * KVH);
    float yr, yi;
    if (t < START) {
        const float* src = cache_k + ((size_t)(b * KV_LEN + t) * KVH + kh) * HD;
        yr = src[2 * p];
        yi = src[2 * p + 1];
    } else {
        int s = t - START;
        const float* src = kf + (size_t)(b * SS + s) * NK + kh * HD;
        float xr = src[2 * p], xi = src[2 * p + 1];
        float inv = powf(10000.0f, -(float)(2 * p) / (float)HD);
        float ang = (float)s * inv;
        float sn, cs;
        sincosf(ang, &sn, &cs);
        yr = xr * cs - xi * sn;
        yi = xr * sn + xi * cs;
    }
    unsigned short* dst = Kbf + (size_t)((b * KVH + kh) * TT + t) * HD;
    dst[2 * p]     = f2bf(yr);
    dst[2 * p + 1] = f2bf(yi);
}

__global__ void build_v_kernel(const float* __restrict__ cache_v,
                               const float* __restrict__ vf,
                               unsigned short* __restrict__ Vt) {
    int i = blockIdx.x * blockDim.x + threadIdx.x;
    const int total = BB * KVH * TT * HD;
    if (i >= total) return;
    int d  = i % HD;
    int t  = (i / HD) % TT;
    int kh = (i / (HD * TT)) % KVH;
    int b  = i / (HD * TT * KVH);
    float v = (t < START)
        ? cache_v[((size_t)(b * KV_LEN + t) * KVH + kh) * HD + d]
        : vf[(size_t)(b * SS + (t - START)) * NK + kh * HD + d];
    Vt[(size_t)((b * KVH + kh) * HD + d) * TT + t] = f2bf(v);
}

// ---------------------------------------------------------------- flash attention
// one wave handles 16 query rows of one (b, head); blockDim = 128 (4 waves)
__global__ void __launch_bounds__(128)
attn_kernel(const unsigned short* __restrict__ Qbf,
            const unsigned short* __restrict__ Kbf,
            const unsigned short* __restrict__ Vt,
            unsigned short* __restrict__ Obf) {
    __shared__ __align__(16) unsigned short plds[4][16 * 32];

    const int lane = threadIdx.x & 31;
    const int wib  = threadIdx.x >> 5;
    const int gw   = blockIdx.x * 4 + wib;
    const int st   = gw & 63;           // q tile (SS/16 = 64)
    const int h    = (gw >> 6) & 31;    // q head
    const int b    = gw >> 11;          // batch
    const int kh   = h >> 2;            // kv head (G = 4)
    const int m0   = st * 16;
    const int l15  = lane & 15;
    const int hiA  = (lane >> 4) * 8;
    const int hiB  = (lane >> 4) * 16;
    const float scale = 0.08838834764831845f;   // 1/sqrt(128)

    // resident Q fragments (16 rows x 128 dims = 4 K-chunks)
    Frag qf[4];
    const unsigned short* qrow = Qbf + (size_t)((b * QH + h) * SS + m0 + l15) * HD;
#pragma unroll
    for (int c = 0; c < 4; ++c) {
        qf[c].u[0] = *(const v4u*)(qrow + c * 32 + hiA);
        qf[c].u[1] = *(const v4u*)(qrow + c * 32 + hiA + 16);
    }

    const unsigned short* kbase = Kbf + (size_t)(b * KVH + kh) * TT * HD;
    const unsigned short* vbase = Vt  + (size_t)(b * KVH + kh) * HD * TT;

    v8f acc[8];
#pragma unroll
    for (int j = 0; j < 8; ++j)
#pragma unroll
        for (int r = 0; r < 8; ++r) acc[j][r] = 0.0f;

    float mstate[8], lstate[8];
#pragma unroll
    for (int r = 0; r < 8; ++r) { mstate[r] = -__builtin_inff(); lstate[r] = 0.0f; }

    unsigned short* myp = &plds[wib][0];

    for (int t0 = 0; t0 < TT; t0 += 32) {
        const unsigned short* kr1 = kbase + (size_t)(t0 + l15) * HD;
        const unsigned short* kr2 = kbase + (size_t)(t0 + 16 + l15) * HD;

        // ---- V fragments first: their loads fly during scores + softmax
        Frag vb[8];
#pragma unroll
        for (int j = 0; j < 8; ++j) {
            const unsigned short* vr =
                vbase + (size_t)(j * 16 + l15) * TT + t0 + hiA;
            vb[j].u[0] = *(const v4u*)(vr);
            vb[j].u[1] = *(const v4u*)(vr + 16);
        }

        // ---- prefetch next key block
        if (t0 + 32 < TT) {
            __builtin_prefetch(kbase + (size_t)(t0 + 32 + l15) * HD, 0, 3);
            __builtin_prefetch(kbase + (size_t)(t0 + 48 + l15) * HD, 0, 3);
        }

        // ---- scores: double-buffered K fragments over 4 HD-chunks
        v8f s1, s2;
#pragma unroll
        for (int r = 0; r < 8; ++r) { s1[r] = 0.0f; s2[r] = 0.0f; }
        Frag kb1[2], kb2[2];
        kb1[0].u[0] = *(const v4u*)(kr1 + hiA);
        kb1[0].u[1] = *(const v4u*)(kr1 + hiA + 16);
        kb2[0].u[0] = *(const v4u*)(kr2 + hiA);
        kb2[0].u[1] = *(const v4u*)(kr2 + hiA + 16);
#pragma unroll
        for (int c = 0; c < 4; ++c) {
            if (c < 3) {
                const int cn = (c + 1) & 1;
                kb1[cn].u[0] = *(const v4u*)(kr1 + (c + 1) * 32 + hiA);
                kb1[cn].u[1] = *(const v4u*)(kr1 + (c + 1) * 32 + hiA + 16);
                kb2[cn].u[0] = *(const v4u*)(kr2 + (c + 1) * 32 + hiA);
                kb2[cn].u[1] = *(const v4u*)(kr2 + (c + 1) * 32 + hiA + 16);
            }
            s1 = __builtin_amdgcn_wmma_f32_16x16x32_bf16(
                false, qf[c].v, false, kb1[c & 1].v, (short)0, s1, false, false);
            s2 = __builtin_amdgcn_wmma_f32_16x16x32_bf16(
                false, qf[c].v, false, kb2[c & 1].v, (short)0, s2, false, false);
        }

        // ---- online softmax (rows at vec elem r per lane-half; cols = lane&15)
        float p1[8], p2[8], fac[8];
#pragma unroll
        for (int r = 0; r < 8; ++r) {
            float a = s1[r] * scale;
            float c = s2[r] * scale;
            float mx = fmaxf(a, c);
            mx = fmaxf(mx, __shfl_xor(mx, 1));
            mx = fmaxf(mx, __shfl_xor(mx, 2));
            mx = fmaxf(mx, __shfl_xor(mx, 4));
            mx = fmaxf(mx, __shfl_xor(mx, 8));
            float mnew = fmaxf(mstate[r], mx);
            float e1 = __expf(a - mnew);
            float e2 = __expf(c - mnew);
            float f  = __expf(mstate[r] - mnew);
            float rs = e1 + e2;
            rs += __shfl_xor(rs, 1);
            rs += __shfl_xor(rs, 2);
            rs += __shfl_xor(rs, 4);
            rs += __shfl_xor(rs, 8);
            lstate[r] = lstate[r] * f + rs;
            mstate[r] = mnew;
            p1[r] = e1; p2[r] = e2; fac[r] = f;
        }
#pragma unroll
        for (int j = 0; j < 8; ++j)
#pragma unroll
            for (int r = 0; r < 8; ++r) acc[j][r] *= fac[r];

        // ---- P: C-layout -> LDS -> A-layout fragment (wave-private tile)
#pragma unroll
        for (int r = 0; r < 8; ++r) {
            const int m = hiA + r;
            myp[m * 32 + l15]      = f2bf(p1[r]);
            myp[m * 32 + 16 + l15] = f2bf(p2[r]);
        }
        Frag pf;
        pf.u[0] = *(const v4u*)(myp + l15 * 32 + hiA);
        pf.u[1] = *(const v4u*)(myp + l15 * 32 + hiA + 16);

        // ---- O += P(16x32) * V(32x128): 8 WMMAs on prefetched V fragments
#pragma unroll
        for (int j = 0; j < 8; ++j) {
            acc[j] = __builtin_amdgcn_wmma_f32_16x16x32_bf16(
                false, pf.v, false, vb[j].v, (short)0, acc[j], false, false);
        }
    }

    // ---- normalize and emit O as bf16 rows of (B*S, EMB)
#pragma unroll
    for (int j = 0; j < 8; ++j)
#pragma unroll
        for (int r = 0; r < 8; ++r) {
            const int m = m0 + hiA + r;
            const float val = acc[j][r] / lstate[r];
            Obf[(size_t)(b * SS + m) * EMB + h * HD + j * 16 + l15] = f2bf(val);
        }
}

// ---------------------------------------------------------------- launcher
extern "C" void kernel_launch(void* const* d_in, const int* in_sizes, int n_in,
                              void* d_out, int out_size, void* d_ws, size_t ws_size,
                              hipStream_t stream) {
    (void)in_sizes; (void)n_in; (void)out_size; (void)ws_size;
    const float* x  = (const float*)d_in[0];
    const float* ck = (const float*)d_in[1];
    const float* cv = (const float*)d_in[2];
    const float* Wq = (const float*)d_in[3];
    const float* Wk = (const float*)d_in[4];
    const float* Wv = (const float*)d_in[5];
    const float* Wo = (const float*)d_in[6];
    const float* bo = (const float*)d_in[7];
    float* out = (float*)d_out;

    char* ws = (char*)d_ws;
    size_t off = 0;
    auto take = [&](size_t bytes) -> char* {
        char* p = ws + off;
        off += (bytes + 255) & ~(size_t)255;
        return p;
    };

    unsigned short* xbf  = (unsigned short*)take((size_t)MROWS * EMB * 2); // reused as Qbf
    unsigned short* wqT  = (unsigned short*)take((size_t)EMB * EMB * 2);
    unsigned short* wkT  = (unsigned short*)take((size_t)NK * EMB * 2);
    unsigned short* wvT  = (unsigned short*)take((size_t)NK * EMB * 2);
    unsigned short* woT  = (unsigned short*)take((size_t)EMB * EMB * 2);
    float*          qf32 = (float*)take((size_t)MROWS * EMB * 4);          // reused as Obf
    float*          kf32 = (float*)take((size_t)MROWS * NK * 4);
    float*          vf32 = (float*)take((size_t)MROWS * NK * 4);
    unsigned short* Kbf  = (unsigned short*)take((size_t)BB * KVH * TT * HD * 2);
    unsigned short* Vt   = (unsigned short*)take((size_t)BB * KVH * TT * HD * 2);
    unsigned short* Qbf  = xbf;                       // alias: x done after QKV GEMMs
    unsigned short* Obf  = (unsigned short*)qf32;     // alias: qf32 done after RoPE

    // 1) bf16 conversions (weights transposed)
    {
        int n = MROWS * EMB;
        f32_to_bf16_kernel<<<(n + 255) / 256, 256, 0, stream>>>(x, xbf, n);
    }
    transpose_to_bf16_kernel<<<(EMB * EMB + 255) / 256, 256, 0, stream>>>(Wq, wqT, EMB, EMB);
    transpose_to_bf16_kernel<<<(EMB * NK + 255) / 256, 256, 0, stream>>>(Wk, wkT, EMB, NK);
    transpose_to_bf16_kernel<<<(EMB * NK + 255) / 256, 256, 0, stream>>>(Wv, wvT, EMB, NK);
    transpose_to_bf16_kernel<<<(EMB * EMB + 255) / 256, 256, 0, stream>>>(Wo, woT, EMB, EMB);

    // 2) QKV projections (grid.y = M/64 now)
    gemm_bf16_wmma_kernel<<<dim3(EMB / 256, MROWS / 64), 256, 0, stream>>>(
        xbf, wqT, qf32, nullptr, MROWS, EMB, EMB);
    gemm_bf16_wmma_kernel<<<dim3(NK / 256, MROWS / 64), 256, 0, stream>>>(
        xbf, wkT, kf32, nullptr, MROWS, NK, EMB);
    gemm_bf16_wmma_kernel<<<dim3(NK / 256, MROWS / 64), 256, 0, stream>>>(
        xbf, wvT, vf32, nullptr, MROWS, NK, EMB);

    // 3) RoPE + KV cache merge (bf16 operand layouts for attention)
    {
        int n = BB * SS * QH * (HD / 2);
        rope_q_kernel<<<(n + 255) / 256, 256, 0, stream>>>(qf32, Qbf);
    }
    {
        int n = BB * KVH * TT * (HD / 2);
        build_k_kernel<<<(n + 255) / 256, 256, 0, stream>>>(ck, kf32, Kbf);
    }
    {
        int n = BB * KVH * TT * HD;
        build_v_kernel<<<(n + 255) / 256, 256, 0, stream>>>(cv, vf32, Vt);
    }

    // 4) WMMA flash attention: B*QH*(S/16) = 4096 waves, 4 waves/block
    attn_kernel<<<(BB * QH * (SS / 16)) / 4, 128, 0, stream>>>(Qbf, Kbf, Vt, Obf);

    // 5) output projection + bias -> fp32 d_out
    gemm_bf16_wmma_kernel<<<dim3(EMB / 256, MROWS / 64), 256, 0, stream>>>(
        Obf, woT, out, bo, MROWS, EMB, EMB);
}
